// SimpleCNN_82849919140408
// MI455X (gfx1250) — matrix operand order repaired and checked by reading, not compiled, not run on hardware
//
#include <hip/hip_runtime.h>
#include <hip/hip_bf16.h>
#include <stdint.h>

// ---------------------------------------------------------------------------
// MI455X (gfx1250) implementation.
//  - wave32; WMMA f32_16x16x32_f16 for conv2 (implicit GEMM) and fc
//  - conv2 input slab DMA'd into LDS by the Tensor Data Mover with pad-based
//    x-halo insertion; B fragments staged in LDS -> inner loop is ds+wmma only
//  - fused BN(+stats)/ReLU/softmax-basis-pool elementwise kernels
// ---------------------------------------------------------------------------

typedef __attribute__((ext_vector_type(16))) _Float16     v16h;
typedef __attribute__((ext_vector_type(8)))  float        v8f;
typedef __attribute__((ext_vector_type(4)))  unsigned int u32x4;
typedef __attribute__((ext_vector_type(4)))  unsigned int su4;
typedef __attribute__((ext_vector_type(8)))  unsigned int su8;

union AFrag { v16h v; u32x4 q[2]; };

#define USE_TDM 1

// ---------------- workspace layout (bytes) ----------------
// y1   : conv1 output,  [1024,16,32,32] f32           =  67,108,864
// y2   : conv2 output,  NHWC [1024,16,16,32] f32      =  33,554,432
// h1p  : pooled1,       NHWC [1024,16,16,32] f16      =  16,777,216
// h2p  : pooled2 flat,  [1024,4096] f16               =   8,388,608
// w2f  : conv2 weight fragments [9][2][32][16] f16    =      18,432
// fcf  : fc weight fragments   [128][32][16] f16      =     131,072
// stats: 192 f32
#define Y1_OFF    0ull
#define Y2_OFF    67108864ull
#define H1P_OFF   100663296ull
#define H2P_OFF   117440512ull
#define W2F_OFF   125829120ull
#define FCF_OFF   125847552ull
#define STATS_OFF 125978624ull

// ===========================================================================
// K1: conv1 3x3 (3->16), bias, NCHW f32 out, fused BN1 partial stats.
// ===========================================================================
__global__ void conv1_bnstat(const float* __restrict__ x,
                             const float* __restrict__ w,
                             const float* __restrict__ b,
                             float* __restrict__ y1,
                             float* __restrict__ gsum, float* __restrict__ gsq) {
    __shared__ float sw[28];
    __shared__ float rs[256], rq[256];
    const int n  = blockIdx.x >> 4;
    const int oc = blockIdx.x & 15;
    const int tid = threadIdx.x;
    if (tid < 27) sw[tid] = w[oc * 27 + tid];
    if (tid == 27) sw[27] = b[oc];
    __syncthreads();
    const float* xin  = x + (size_t)n * 3 * 1024;
    float*       yout = y1 + (size_t)blockIdx.x * 1024;
    float ls = 0.f, lq = 0.f;
    for (int t = 0; t < 4; ++t) {
        const int p  = tid + t * 256;
        const int yy = p >> 5, xx = p & 31;
        float acc = sw[27];
        #pragma unroll
        for (int ic = 0; ic < 3; ++ic) {
            #pragma unroll
            for (int ky = 0; ky < 3; ++ky) {
                const int yi = yy + ky - 1;
                if (yi < 0 || yi > 31) continue;
                #pragma unroll
                for (int kx = 0; kx < 3; ++kx) {
                    const int xi = xx + kx - 1;
                    if (xi < 0 || xi > 31) continue;
                    acc += xin[ic * 1024 + yi * 32 + xi] * sw[ic * 9 + ky * 3 + kx];
                }
            }
        }
        yout[p] = acc;
        ls += acc; lq += acc * acc;
    }
    rs[tid] = ls; rq[tid] = lq;
    __syncthreads();
    for (int s = 128; s > 0; s >>= 1) {
        if (tid < s) { rs[tid] += rs[tid + s]; rq[tid] += rq[tid + s]; }
        __syncthreads();
    }
    if (tid == 0) { atomicAdd(&gsum[oc], rs[0]); atomicAdd(&gsq[oc], rq[0]); }
}

// ===========================================================================
// K2: BN finalize
// ===========================================================================
__global__ void bn_finalize(const float* __restrict__ sum, const float* __restrict__ sq,
                            const float* __restrict__ gamma, const float* __restrict__ beta,
                            float* __restrict__ scale, float* __restrict__ bias,
                            int C, float inv_count) {
    const int c = threadIdx.x;
    if (c < C) {
        const float mean = sum[c] * inv_count;
        const float var  = sq[c] * inv_count - mean * mean;
        const float s    = gamma[c] * rsqrtf(var + 1e-5f);
        scale[c] = s;
        bias[c]  = beta[c] - mean * s;
    }
}

// ===========================================================================
// K3: BN1 + ReLU + basis_pool(K=2, temp=0.1) -> h1p NHWC f16 (C = c*2+k, 32)
// ===========================================================================
__global__ void pool1_kernel(const float* __restrict__ y1,
                             const float* __restrict__ scale, const float* __restrict__ bias,
                             const float* __restrict__ pw, const float* __restrict__ pb,
                             _Float16* __restrict__ h1p) {
    const int idx = blockIdx.x * 256 + threadIdx.x;      // 4,194,304 total
    const int pix = idx & 255;
    const int oy = pix >> 4, ox = pix & 15;
    const int nc = idx >> 8;
    const int c = nc & 15, n = nc >> 4;
    const float* plane = y1 + (size_t)nc * 1024;
    const float sc = scale[c], bi = bias[c];
    float a[4];
    #pragma unroll
    for (int dy = 0; dy < 2; ++dy)
        #pragma unroll
        for (int dx = 0; dx < 2; ++dx) {
            const float v = plane[(2 * oy + dy) * 32 + (2 * ox + dx)];
            a[dy * 2 + dx] = fmaxf(v * sc + bi, 0.0f);
        }
    _Float16* outp = h1p + (((size_t)(n * 16 + oy) * 16 + ox) * 32 + c * 2);
    #pragma unroll
    for (int k = 0; k < 2; ++k) {
        const float w0 = pw[k], b0 = pb[k];
        const float s0 = (a[0] * w0 + b0) * 10.f, s1 = (a[1] * w0 + b0) * 10.f;
        const float s2 = (a[2] * w0 + b0) * 10.f, s3 = (a[3] * w0 + b0) * 10.f;
        const float mx = fmaxf(fmaxf(s0, s1), fmaxf(s2, s3));
        const float e0 = __expf(s0 - mx), e1 = __expf(s1 - mx);
        const float e2 = __expf(s2 - mx), e3 = __expf(s3 - mx);
        const float num = e0 * a[0] + e1 * a[1] + e2 * a[2] + e3 * a[3];
        outp[k] = (_Float16)(num / (e0 + e1 + e2 + e3));
    }
}

// ===========================================================================
// K4a: swizzle conv2 weights -> per-lane WMMA B fragments
// frag[tap][ntile][lane][16] f16; B-layout: N = lane&15, K = 8*(lane>>4)+(j&7)+16*(j>>3)
// ===========================================================================
__global__ void prep_w2frag(const float* __restrict__ w2, _Float16* __restrict__ frag) {
    const int idx = blockIdx.x * 256 + threadIdx.x;      // 9216 total
    if (idx >= 9 * 2 * 32 * 16) return;
    const int j    = idx & 15;
    const int lane = (idx >> 4) & 31;
    const int tile = (idx >> 9) & 1;
    const int tap  = idx >> 10;
    const int oc = tile * 16 + (lane & 15);
    const int ic = (lane >> 4) * 8 + (j & 7) + ((j >> 3) << 4);
    const int ty = tap / 3, tx = tap % 3;
    frag[idx] = (_Float16)w2[((oc * 32 + ic) * 3 + ty) * 3 + tx];
}

// ===========================================================================
// K4b: swizzle fc weights [10,4096] -> frag[128 kchunk][32 lane][16] f16
// ===========================================================================
__global__ void prep_fcfrag(const float* __restrict__ fw, _Float16* __restrict__ frag) {
    const int idx = blockIdx.x * 256 + threadIdx.x;      // 65536 total
    const int j    = idx & 15;
    const int lane = (idx >> 4) & 31;
    const int kc   = idx >> 9;
    const int o = lane & 15;
    const int k = kc * 32 + (lane >> 4) * 8 + (j & 7) + ((j >> 3) << 4);
    frag[idx] = (o < 10) ? (_Float16)fw[o * 4096 + k] : (_Float16)0.0f;
}

// ===========================================================================
// K5: conv2 3x3 (32->32) implicit-GEMM WMMA.
// Block = 8 rows of one image. Input slab (9 valid rows + zero halos) is
// DMA'd into LDS by the TDM with pad-based x-halo insertion:
//   LDS row stride = 1024B data + 128B pad = 1152B; real pixels start at +64B,
//   so column slots -1 and 16 are the pre-zeroed pad/halo bytes.
// Inner loop: unconditional ds_load_b128 A/B + v_wmma_f32_16x16x32_f16.
// ===========================================================================
__global__ void conv2_wmma(const _Float16* __restrict__ h1p,
                           const _Float16* __restrict__ w2frag,
                           const float* __restrict__ conv2_b,
                           float* __restrict__ y2,
                           float* __restrict__ gsum, float* __restrict__ gsq) {
    __shared__ __align__(16) unsigned int sInU[2880];   // 10 rows * 1152 B
    __shared__ __align__(16) unsigned int sWU[4608];    // 18432 B of B fragments
    __shared__ float sS[32], sQ[32];

    const int tid = threadIdx.x;
    const int n  = blockIdx.x >> 1;
    const int y0 = (blockIdx.x & 1) * 8;

    // phase 0: zero input staging (covers halo rows + pad columns), stage B
    #pragma unroll
    for (int i = 0; i < 12; ++i) {
        const int k = tid + i * 256;
        if (k < 2880) sInU[k] = 0u;
    }
    #pragma unroll
    for (int i = 0; i < 18; ++i)
        sWU[tid + i * 256] = ((const unsigned int*)w2frag)[tid + i * 256];
    if (tid < 32) { sS[tid] = 0.f; sQ[tid] = 0.f; }
    __syncthreads();

    // phase 1: TDM DMA of the 9 valid input rows into LDS
    if (tid == 0) {
        const int ystart = (y0 == 0) ? 0 : 7;            // first valid global row
        const int r0     = (y0 == 0) ? 1 : 0;            // its LDS row slot
        const unsigned lds_addr =
            (unsigned)(size_t)(void*)sInU + 64u + (unsigned)r0 * 1152u;
        const unsigned long long ga =
            (unsigned long long)(size_t)(h1p + ((size_t)(n * 16 + ystart) * 512));
        const unsigned units = 9u * 128u;                // 9 rows * 1024B in 8B units
        su4 g0; su8 g1;
        g0[0] = 1u;                                       // count=1 (valid D#)
        g0[1] = lds_addr;
        g0[2] = (unsigned)(ga & 0xffffffffu);
        g0[3] = (unsigned)((ga >> 32) & 0x01ffffffu) | (2u << 30);  // type=2
        // data_size=8B | pad_enable | pad_interval=256 dw | pad_amount=32 dw
        g1[0] = (3u << 16) | (1u << 20) | (7u << 22) | (31u << 25);
        g1[1] = (units & 0xffffu) << 16;                  // tensor_dim0 lo
        g1[2] = (units >> 16) | (1u << 16);               // tensor_dim0 hi | tensor_dim1=1
        g1[3] = (units & 0xffffu) << 16;                  // tile_dim0
        g1[4] = 0u;                                       // tile_dim1/2 unused
        g1[5] = units;                                    // tensor_dim0_stride lo
        g1[6] = 0u;
        g1[7] = 0u;
#if USE_TDM
        asm volatile("tensor_load_to_lds %0, %1" :: "s"(g0), "s"(g1) : "memory");
        __builtin_amdgcn_s_wait_tensorcnt(0);
#endif
    }
    __syncthreads();

    const int lane = tid & 31;
    const int wv   = tid >> 5;
    const int y    = y0 + wv;
    const int half = lane >> 4, m = lane & 15;
    const int h8   = half * 8;
    const _Float16* al = (const _Float16*)sInU;
    const _Float16* wl = (const _Float16*)sWU;

    v8f c0 = {}, c1 = {};
    #pragma unroll
    for (int ty = 0; ty < 3; ++ty) {
        #pragma unroll
        for (int tx = 0; tx < 3; ++tx) {
            const int tap = ty * 3 + tx;
            AFrag a, b0, b1;
            // LDS row slot (wv+ty) covers global row y+ty-1; column slot m+tx
            const _Float16* p = al + (wv + ty) * 576 + (m + tx) * 32 + h8;
            a.q[0] = *(const u32x4*)(p);
            a.q[1] = *(const u32x4*)(p + 16);
            const _Float16* wp0 = wl + ((tap * 2 + 0) * 32 + lane) * 16;
            const _Float16* wp1 = wl + ((tap * 2 + 1) * 32 + lane) * 16;
            b0.q[0] = *(const u32x4*)(wp0); b0.q[1] = *(const u32x4*)(wp0 + 8);
            b1.q[0] = *(const u32x4*)(wp1); b1.q[1] = *(const u32x4*)(wp1 + 8);
            c0 = __builtin_amdgcn_wmma_f32_16x16x32_f16(false, a.v, false, b0.v,
                                                        (short)0, c0, false, false);
            c1 = __builtin_amdgcn_wmma_f32_16x16x32_f16(false, a.v, false, b1.v,
                                                        (short)0, c1, false, false);
        }
    }

    // D layout: lane holds column N = m; VGPR r -> pixel M = 8*half + r.
    const int oc0 = m, oc1 = 16 + m;
    const float bb0 = conv2_b[oc0], bb1 = conv2_b[oc1];
    float s0 = 0.f, q0 = 0.f, s1 = 0.f, q1 = 0.f;
    float* outrow = y2 + ((size_t)(n * 16 + y) * 16) * 32;
    #pragma unroll
    for (int r = 0; r < 8; ++r) {
        const int mm = h8 + r;
        const float v0 = c0[r] + bb0;
        const float v1 = c1[r] + bb1;
        outrow[mm * 32 + oc0] = v0;
        outrow[mm * 32 + oc1] = v1;
        s0 += v0; q0 += v0 * v0;
        s1 += v1; q1 += v1 * v1;
    }
    atomicAdd(&sS[oc0], s0); atomicAdd(&sQ[oc0], q0);
    atomicAdd(&sS[oc1], s1); atomicAdd(&sQ[oc1], q1);
    __syncthreads();
    if (tid < 32) {
        atomicAdd(&gsum[tid], sS[tid]);
        atomicAdd(&gsq[tid],  sQ[tid]);
    }
}

// ===========================================================================
// K6: BN2 + ReLU + basis_pool2 -> h2p f16 [1024,4096], f = (c*2+k)*64+oy*8+ox
// ===========================================================================
__global__ void pool2_kernel(const float* __restrict__ y2,
                             const float* __restrict__ scale, const float* __restrict__ bias,
                             const float* __restrict__ pw, const float* __restrict__ pb,
                             _Float16* __restrict__ h2p) {
    const int idx = blockIdx.x * 256 + threadIdx.x;      // 2,097,152 total
    const int pix = idx & 63;
    const int oy = pix >> 3, ox = pix & 7;
    const int nc = idx >> 6;
    const int c = nc & 31, n = nc >> 5;
    const float sc = scale[c], bi = bias[c];
    float a[4];
    #pragma unroll
    for (int dy = 0; dy < 2; ++dy)
        #pragma unroll
        for (int dx = 0; dx < 2; ++dx) {
            const float v = y2[(((size_t)n * 16 + 2 * oy + dy) * 16 + (2 * ox + dx)) * 32 + c];
            a[dy * 2 + dx] = fmaxf(v * sc + bi, 0.0f);
        }
    _Float16* outp = h2p + (size_t)n * 4096 + (c * 2) * 64 + oy * 8 + ox;
    #pragma unroll
    for (int k = 0; k < 2; ++k) {
        const float w0 = pw[k], b0 = pb[k];
        const float s0 = (a[0] * w0 + b0) * 10.f, s1 = (a[1] * w0 + b0) * 10.f;
        const float s2 = (a[2] * w0 + b0) * 10.f, s3 = (a[3] * w0 + b0) * 10.f;
        const float mx = fmaxf(fmaxf(s0, s1), fmaxf(s2, s3));
        const float e0 = __expf(s0 - mx), e1 = __expf(s1 - mx);
        const float e2 = __expf(s2 - mx), e3 = __expf(s3 - mx);
        const float num = e0 * a[0] + e1 * a[1] + e2 * a[2] + e3 * a[3];
        outp[(size_t)k * 64] = (_Float16)(num / (e0 + e1 + e2 + e3));
    }
}

// ===========================================================================
// K7: fc via WMMA: out[1024,10] = h2p[1024,4096] @ W^T + b.
// 64 waves; each wave: M-tile = 16 batch rows, N = 16 (10 valid), K = 4096.
// ===========================================================================
__global__ void fc_wmma(const _Float16* __restrict__ h2p,
                        const _Float16* __restrict__ fcfrag,
                        const float* __restrict__ fc_b,
                        float* __restrict__ out) {
    const int lane = threadIdx.x & 31;
    const int g = blockIdx.x * 8 + (threadIdx.x >> 5);   // M tile 0..63
    const int half = lane >> 4, m = lane & 15;
    const int h8 = half * 8;
    v8f c = {};
    const _Float16* arow = h2p + (size_t)(g * 16 + m) * 4096;
    for (int kc = 0; kc < 128; ++kc) {
        AFrag a, b;
        const _Float16* p = arow + kc * 32;
        __builtin_prefetch((const void*)(p + 1024), 0, 0);  // ~32 iters ahead
        a.q[0] = *(const u32x4*)(p + h8);
        a.q[1] = *(const u32x4*)(p + 16 + h8);
        const _Float16* wp = fcfrag + ((size_t)kc * 32 + lane) * 16;
        b.q[0] = *(const u32x4*)(wp);
        b.q[1] = *(const u32x4*)(wp + 8);
        c = __builtin_amdgcn_wmma_f32_16x16x32_f16(false, a.v, false, b.v,
                                                   (short)0, c, false, false);
    }
    if (m < 10) {
        const float bias = fc_b[m];
        #pragma unroll
        for (int r = 0; r < 8; ++r)
            out[(size_t)(g * 16 + h8 + r) * 10 + m] = c[r] + bias;
    }
}

// ===========================================================================
extern "C" void kernel_launch(void* const* d_in, const int* in_sizes, int n_in,
                              void* d_out, int out_size, void* d_ws, size_t ws_size,
                              hipStream_t stream) {
    (void)in_sizes; (void)n_in; (void)out_size; (void)ws_size;
    const float* x        = (const float*)d_in[0];
    const float* conv1_w  = (const float*)d_in[1];
    const float* conv1_b  = (const float*)d_in[2];
    const float* bn1_g    = (const float*)d_in[3];
    const float* bn1_b    = (const float*)d_in[4];
    const float* proj1_w  = (const float*)d_in[5];
    const float* proj1_b  = (const float*)d_in[6];
    const float* conv2_w  = (const float*)d_in[7];
    const float* conv2_b  = (const float*)d_in[8];
    const float* bn2_g    = (const float*)d_in[9];
    const float* bn2_b    = (const float*)d_in[10];
    const float* proj2_w  = (const float*)d_in[11];
    const float* proj2_b  = (const float*)d_in[12];
    const float* fc_w     = (const float*)d_in[13];
    const float* fc_b     = (const float*)d_in[14];
    float* out = (float*)d_out;

    char* ws = (char*)d_ws;
    float*    y1     = (float*)(ws + Y1_OFF);
    float*    y2     = (float*)(ws + Y2_OFF);
    _Float16* h1p    = (_Float16*)(ws + H1P_OFF);
    _Float16* h2p    = (_Float16*)(ws + H2P_OFF);
    _Float16* w2frag = (_Float16*)(ws + W2F_OFF);
    _Float16* fcfrag = (_Float16*)(ws + FCF_OFF);
    float*    stats  = (float*)(ws + STATS_OFF);
    float *sum1 = stats,      *sq1 = stats + 16, *scale1 = stats + 32, *bias1 = stats + 48;
    float *sum2 = stats + 64, *sq2 = stats + 96, *scale2 = stats + 128, *bias2 = stats + 160;

    hipMemsetAsync(stats, 0, 192 * sizeof(float), stream);

    prep_w2frag<<<36, 256, 0, stream>>>(conv2_w, w2frag);
    prep_fcfrag<<<256, 256, 0, stream>>>(fc_w, fcfrag);

    conv1_bnstat<<<16384, 256, 0, stream>>>(x, conv1_w, conv1_b, y1, sum1, sq1);
    bn_finalize<<<1, 32, 0, stream>>>(sum1, sq1, bn1_g, bn1_b, scale1, bias1,
                                      16, 1.0f / (1024.0f * 32.0f * 32.0f));
    pool1_kernel<<<16384, 256, 0, stream>>>(y1, scale1, bias1, proj1_w, proj1_b, h1p);

    conv2_wmma<<<2048, 256, 0, stream>>>(h1p, w2frag, conv2_b, y2, sum2, sq2);
    bn_finalize<<<1, 32, 0, stream>>>(sum2, sq2, bn2_g, bn2_b, scale2, bias2,
                                      32, 1.0f / (1024.0f * 16.0f * 16.0f));
    pool2_kernel<<<8192, 256, 0, stream>>>(y2, scale2, bias2, proj2_w, proj2_b, h2p);

    fc_wmma<<<8, 256, 0, stream>>>(h2p, fcfrag, fc_b, out);
}
